// MultiHeadAttention_78692390797835
// MI455X (gfx1250) — matrix operand order repaired and checked by reading, not compiled, not run on hardware
//
#include <hip/hip_runtime.h>

// ---------------- CDNA5 WMMA types ----------------
typedef __attribute__((ext_vector_type(16))) __bf16 v16bf;
typedef __attribute__((ext_vector_type(8)))  float  v8f;
typedef __attribute__((ext_vector_type(4)))  __bf16 v4bf;
typedef __attribute__((__vector_size__(16))) int    v4i_t;

union FragB16 { v16bf v; uint4 q[2]; };

static __device__ __forceinline__ v8f zero8() {
  v8f z;
#pragma unroll
  for (int i = 0; i < 8; ++i) z[i] = 0.0f;
  return z;
}

#define WMMA_BF16(a, b, c) \
  __builtin_amdgcn_wmma_f32_16x16x32_bf16(false, (a), false, (b), (short)0, (c), false, false)

// ---- gfx1250 async global->LDS copy (ASYNCcnt path), with safe fallback ----
#if defined(__AMDGCN__) && __has_builtin(__builtin_amdgcn_global_load_async_to_lds_b128)
#define ASYNC_LDS 1
#else
#define ASYNC_LDS 0
#endif

static __device__ __forceinline__ void cp16_g2l(const __bf16* g, __bf16* l) {
#if ASYNC_LDS
  __builtin_amdgcn_global_load_async_to_lds_b128(
      (__attribute__((address_space(1))) v4i_t*)g,
      (__attribute__((address_space(3))) v4i_t*)l, 0, 0);
#else
  *(uint4*)l = *(const uint4*)g;
#endif
}
static __device__ __forceinline__ void async_fence() {
#if ASYNC_LDS
  asm volatile("s_wait_asynccnt 0" ::: "memory");
#endif
}

// ---------------- problem constants ----------------
#define BSZ    4
#define SEQ    1024
#define DMODEL 4096
#define NH     32
#define NKV    8
#define HD     128
#define PASTN  1024
#define TOTK   2048

// =====================================================================
// GEMM:  C[M,N] = A[M,K] @ W[N,K]^T   (bf16 in, f32 accumulate)
// 256 threads = 8 waves; block tile 128x128; wave tile 32x64; Kc=64
// (16 WMMA + 24 ds_load_b128 per barrier); double-buffered LDS fed by
// async global->LDS b128 copies. N,K compile-time.
// =====================================================================
template <int N, int K, bool OUT_BF16>
__global__ __launch_bounds__(256)
void gemm_bf16(const __bf16* __restrict__ A, const __bf16* __restrict__ W,
               void* __restrict__ Cout) {
  constexpr int BM = 128, BN = 128, KC = 64, LDSS = KC + 8; // 144B rows: 16B-aligned
  __shared__ __bf16 As[2][BM * LDSS];
  __shared__ __bf16 Bs[2][BN * LDSS];

  const int tid  = threadIdx.x;
  const int lane = tid & 31;
  const int w    = tid >> 5;
  const int wm   = w & 3;          // 4 wave rows (32 rows each)
  const int wn   = w >> 2;         // 2 wave cols (64 cols each)
  const int bm   = blockIdx.y * BM;
  const int bn   = blockIdx.x * BN;
  const int l16  = lane & 15;
  const int hf   = lane >> 4;
  const int c0   = hf * 8;

  v8f acc[2][4];
#pragma unroll
  for (int mi = 0; mi < 2; ++mi)
#pragma unroll
    for (int ni = 0; ni < 4; ++ni) acc[mi][ni] = zero8();

  // staging: 1024 16B chunks per 128x64 tile / 256 threads = 4 chunks each
  auto stage = [&](int k0, int buf) {
#pragma unroll
    for (int i = 0; i < 4; ++i) {
      int idx = tid + i * 256;
      int r = idx >> 3, c = (idx & 7) * 8;
      cp16_g2l(A + (size_t)(bm + r) * K + k0 + c, &As[buf][r * LDSS + c]);
      cp16_g2l(W + (size_t)(bn + r) * K + k0 + c, &Bs[buf][r * LDSS + c]);
    }
  };

  stage(0, 0);
  async_fence();
  __syncthreads();

  int buf = 0;
  for (int k0 = 0; k0 < K; k0 += KC) {
    if (k0 + KC < K) stage(k0 + KC, buf ^ 1);   // overlap next stage with compute

#pragma unroll
    for (int kk = 0; kk < KC; kk += 32) {
      FragB16 af[2], bfr[4];
#pragma unroll
      for (int mi = 0; mi < 2; ++mi) {
        const __bf16* p = &As[buf][(wm * 32 + mi * 16 + l16) * LDSS + kk];
        af[mi].q[0] = *(const uint4*)(p + c0);
        af[mi].q[1] = *(const uint4*)(p + 16 + c0);
      }
#pragma unroll
      for (int ni = 0; ni < 4; ++ni) {
        const __bf16* p = &Bs[buf][(wn * 64 + ni * 16 + l16) * LDSS + kk];
        bfr[ni].q[0] = *(const uint4*)(p + c0);
        bfr[ni].q[1] = *(const uint4*)(p + 16 + c0);
      }
#pragma unroll
      for (int mi = 0; mi < 2; ++mi)
#pragma unroll
        for (int ni = 0; ni < 4; ++ni)
          acc[mi][ni] = WMMA_BF16(af[mi].v, bfr[ni].v, acc[mi][ni]);
    }

    async_fence();
    __syncthreads();
    buf ^= 1;
  }

  // epilogue: C/D layout N = lane%16, M = r + 8*half
#pragma unroll
  for (int mi = 0; mi < 2; ++mi)
#pragma unroll
    for (int ni = 0; ni < 4; ++ni) {
      const int col  = bn + wn * 64 + ni * 16 + l16;
      const int rowb = bm + wm * 32 + mi * 16 + hf * 8;
      if constexpr (OUT_BF16) {
        __bf16* C = (__bf16*)Cout;
#pragma unroll
        for (int r = 0; r < 8; ++r)
          C[(size_t)(rowb + r) * N + col] = (__bf16)acc[mi][ni][r];
      } else {
        float* C = (float*)Cout;
#pragma unroll
        for (int r = 0; r < 8; ++r)
          C[(size_t)(rowb + r) * N + col] = acc[mi][ni][r];
      }
    }
}

// =====================================================================
// f32 -> bf16 bulk convert (vectorized), one-shot precision downcast
// =====================================================================
__global__ void cvt_f32_bf16_kernel(const float* __restrict__ in,
                                    __bf16* __restrict__ out, int n4) {
  int i = blockIdx.x * blockDim.x + threadIdx.x;
  if (i < n4) {
    float4 f = ((const float4*)in)[i];
    v4bf o = {(__bf16)f.x, (__bf16)f.y, (__bf16)f.z, (__bf16)f.w};
    *(v4bf*)(out + 4 * (size_t)i) = o;
  }
}

// RoPE on Q (fused 1/sqrt(HD)) : qraw[b*S+s, h*128+d] -> qb[b,h,s,d]
__global__ void rope_q_kernel(const __bf16* __restrict__ qraw,
                              const float* __restrict__ cosb,
                              const float* __restrict__ sinb,
                              __bf16* __restrict__ qb) {
  int idx = blockIdx.x * blockDim.x + threadIdx.x;    // B*NH*S*64
  int j = idx & 63;
  int h = (idx >> 6) & 31;
  int s = (idx >> 11) & 1023;
  int b = idx >> 21;
  size_t src = (size_t)(b * SEQ + s) * DMODEL + h * HD + 2 * j;
  float xr = (float)qraw[src], xi = (float)qraw[src + 1];
  float c = cosb[s * 64 + j], sn = sinb[s * 64 + j];
  const float sc = 0.08838834764831845f;  // 1/sqrt(128)
  size_t o = ((size_t)(b * NH + h) * SEQ + s) * HD + 2 * j;
  qb[o]     = (__bf16)((xr * c - xi * sn) * sc);
  qb[o + 1] = (__bf16)((xr * sn + xi * c) * sc);
}

// RoPE on K, appended at position PAST+s : keys[b,kh,2048,128]
__global__ void rope_k_kernel(const __bf16* __restrict__ kraw,
                              const float* __restrict__ cosb,
                              const float* __restrict__ sinb,
                              __bf16* __restrict__ keys) {
  int idx = blockIdx.x * blockDim.x + threadIdx.x;    // B*NKV*S*64
  int j  = idx & 63;
  int kh = (idx >> 6) & 7;
  int s  = (idx >> 9) & 1023;
  int b  = idx >> 19;
  size_t src = (size_t)(b * SEQ + s) * (NKV * HD) + kh * HD + 2 * j;
  float xr = (float)kraw[src], xi = (float)kraw[src + 1];
  float c = cosb[s * 64 + j], sn = sinb[s * 64 + j];
  size_t o = ((size_t)(b * NKV + kh) * TOTK + PASTN + s) * HD + 2 * j;
  keys[o]     = (__bf16)(xr * c - xi * sn);
  keys[o + 1] = (__bf16)(xr * sn + xi * c);
}

// cache_k[b,kh,p,d] f32 -> keys[b,kh,p,d] bf16
__global__ void cache_k_kernel(const float* __restrict__ ck, __bf16* __restrict__ keys) {
  int idx = blockIdx.x * blockDim.x + threadIdx.x;    // B*NKV*PAST*128
  int d  = idx & 127;
  int p  = (idx >> 7) & 1023;
  int kh = (idx >> 17) & 7;
  int b  = idx >> 20;
  keys[((size_t)(b * NKV + kh) * TOTK + p) * HD + d] = (__bf16)ck[idx];
}

// new V bf16 rows [b*S+s, kh*128+d] -> values_t[b,kh,d,PAST+s] (transposed)
__global__ void v_new_kernel(const __bf16* __restrict__ vraw, __bf16* __restrict__ vt) {
  int idx = blockIdx.x * blockDim.x + threadIdx.x;    // B*S*1024
  int d  = idx & 127;
  int kh = (idx >> 7) & 7;
  int s  = (idx >> 10) & 1023;
  int b  = idx >> 20;
  vt[((size_t)(b * NKV + kh) * HD + d) * TOTK + PASTN + s] = vraw[idx];
}

// cache_v[b,kh,p,d] f32 -> values_t[b,kh,d,p]
__global__ void cache_v_kernel(const float* __restrict__ cv, __bf16* __restrict__ vt) {
  int idx = blockIdx.x * blockDim.x + threadIdx.x;    // B*NKV*PAST*128
  int d  = idx & 127;
  int p  = (idx >> 7) & 1023;
  int kh = (idx >> 17) & 7;
  int b  = idx >> 20;
  vt[((size_t)(b * NKV + kh) * HD + d) * TOTK + p] = (__bf16)cv[idx];
}

// =====================================================================
// Flash attention: one wave per (b, h, 16-row Q tile); 64-key chunks
// (4 score tiles) so the ds_bpermute softmax reductions and the P->LDS
// transpose amortize over 2x the keys; online f32 softmax; output bf16.
// =====================================================================
__global__ __launch_bounds__(128)
void attn_kernel(const __bf16* __restrict__ Q, const __bf16* __restrict__ Kk,
                 const __bf16* __restrict__ Vt, __bf16* __restrict__ Oout) {
  __shared__ __bf16 Pt[4][16 * 64];     // 2KB per wave
  const int lane = threadIdx.x & 31;
  const int wave = threadIdx.x >> 5;
  const int tile = blockIdx.x * 4 + wave;  // B*NH*(S/16) tiles
  const int qt  = tile & 63;
  const int h   = (tile >> 6) & 31;
  const int b   = tile >> 11;
  const int kvh = h >> 2;
  const int l16 = lane & 15;
  const int hf  = lane >> 4;
  const int c0  = hf * 8;
  const int q0  = qt * 16;

  FragB16 qf[4];
  const __bf16* qrow = Q + ((size_t)((b * NH + h) * SEQ + q0 + l16)) * HD;
#pragma unroll
  for (int kc = 0; kc < 4; ++kc) {
    qf[kc].q[0] = *(const uint4*)(qrow + kc * 32 + c0);
    qf[kc].q[1] = *(const uint4*)(qrow + kc * 32 + 16 + c0);
  }

  const __bf16* kbase = Kk + (size_t)(b * NKV + kvh) * TOTK * HD;
  const __bf16* vbase = Vt + (size_t)(b * NKV + kvh) * HD * TOTK;

  v8f O[8];
#pragma unroll
  for (int dn = 0; dn < 8; ++dn) O[dn] = zero8();
  float m[8], l[8];
#pragma unroll
  for (int r = 0; r < 8; ++r) { m[r] = -1e30f; l[r] = 0.0f; }

  __bf16* pt = &Pt[wave][0];

  for (int kb = 0; kb < TOTK; kb += 64) {
    // ---- scores: four 16x16 tiles (keys kb+16t, t=0..3) ----
    v8f s[4];
#pragma unroll
    for (int t = 0; t < 4; ++t) s[t] = zero8();
#pragma unroll
    for (int t = 0; t < 4; ++t) {
      const __bf16* kr = kbase + (size_t)(kb + t * 16 + l16) * HD;
#pragma unroll
      for (int kc = 0; kc < 4; ++kc) {
        FragB16 bk;
        int off = kc * 32 + c0;
        bk.q[0] = *(const uint4*)(kr + off);
        bk.q[1] = *(const uint4*)(kr + off + 16);
        s[t] = WMMA_BF16(qf[kc].v, bk.v, s[t]);
      }
    }

    // ---- online softmax over 64 keys (row = r + 8*hf striped on 16 lanes) ----
#pragma unroll
    for (int r = 0; r < 8; ++r) {
      float cm = fmaxf(fmaxf(s[0][r], s[1][r]), fmaxf(s[2][r], s[3][r]));
#pragma unroll
      for (int off = 1; off < 16; off <<= 1)
        cm = fmaxf(cm, __shfl_xor(cm, off, 32));
      float mn   = fmaxf(m[r], cm);
      float corr = __expf(m[r] - mn);
      float p0 = __expf(s[0][r] - mn);
      float p1 = __expf(s[1][r] - mn);
      float p2 = __expf(s[2][r] - mn);
      float p3 = __expf(s[3][r] - mn);
      float rs = (p0 + p1) + (p2 + p3);
#pragma unroll
      for (int off = 1; off < 16; off <<= 1)
        rs += __shfl_xor(rs, off, 32);
      l[r] = l[r] * corr + rs;
      m[r] = mn;
#pragma unroll
      for (int dn = 0; dn < 8; ++dn) O[dn][r] *= corr;
      __bf16* prow = pt + (r + hf * 8) * 64 + l16;
      prow[0]  = (__bf16)p0;
      prow[16] = (__bf16)p1;
      prow[32] = (__bf16)p2;
      prow[48] = (__bf16)p3;
    }
    asm volatile("s_wait_dscnt 0" ::: "memory");   // wave-local LDS RAW fence

    // ---- gather P as two A-fragments (keys 0..31 / 32..63 of the chunk) ----
    FragB16 pf0, pf1;
    pf0.q[0] = *(const uint4*)(pt + l16 * 64 + c0);
    pf0.q[1] = *(const uint4*)(pt + l16 * 64 + 16 + c0);
    pf1.q[0] = *(const uint4*)(pt + l16 * 64 + 32 + c0);
    pf1.q[1] = *(const uint4*)(pt + l16 * 64 + 48 + c0);
    asm volatile("s_wait_dscnt 0" ::: "memory");   // reads done before next iter writes

    // ---- O += P @ V  (V transposed: two 16B k-runs, 32B apart, per B-fragment lane) ----
#pragma unroll
    for (int dn = 0; dn < 8; ++dn) {
      const __bf16* vr = vbase + (size_t)(dn * 16 + l16) * TOTK + kb + c0;
      FragB16 v0, v1;
      v0.q[0] = *(const uint4*)(vr);
      v0.q[1] = *(const uint4*)(vr + 16);
      v1.q[0] = *(const uint4*)(vr + 32);
      v1.q[1] = *(const uint4*)(vr + 48);
      O[dn] = WMMA_BF16(pf0.v, v0.v, O[dn]);
      O[dn] = WMMA_BF16(pf1.v, v1.v, O[dn]);
    }
  }

  float inv[8];
#pragma unroll
  for (int r = 0; r < 8; ++r) inv[r] = 1.0f / l[r];
#pragma unroll
  for (int dn = 0; dn < 8; ++dn) {
    int col = h * HD + dn * 16 + l16;
#pragma unroll
    for (int r = 0; r < 8; ++r) {
      int row = b * SEQ + q0 + r + hf * 8;
      Oout[(size_t)row * DMODEL + col] = (__bf16)(O[dn][r] * inv[r]);
    }
  }
}

// =====================================================================
// Host launch
// =====================================================================
extern "C" void kernel_launch(void* const* d_in, const int* in_sizes, int n_in,
                              void* d_out, int out_size, void* d_ws, size_t ws_size,
                              hipStream_t stream) {
  (void)in_sizes; (void)n_in; (void)out_size; (void)ws_size;
  const float* x    = (const float*)d_in[0];
  const float* fcos = (const float*)d_in[1];
  const float* fsin = (const float*)d_in[2];
  const float* ck   = (const float*)d_in[3];
  const float* cv   = (const float*)d_in[4];
  const float* wq   = (const float*)d_in[5];
  const float* wk   = (const float*)d_in[6];
  const float* wv   = (const float*)d_in[7];
  const float* wo   = (const float*)d_in[8];
  float* out = (float*)d_out;

  char* ws = (char*)d_ws;
  __bf16* xb    = (__bf16*)(ws);                           // 32MB; later attn-out (bf16)
  __bf16* wqb   = (__bf16*)(ws + ((size_t)32  << 20));     // 32MB; later wo (bf16)
  __bf16* wkb   = (__bf16*)(ws + ((size_t)64  << 20));     // 8MB
  __bf16* wvb   = (__bf16*)(ws + ((size_t)72  << 20));     // 8MB
  __bf16* qraw  = (__bf16*)(ws + ((size_t)80  << 20));     // 32MB (Q pre-RoPE)
  __bf16* kraw  = (__bf16*)(ws + ((size_t)112 << 20));     // 8MB
  __bf16* vraw  = (__bf16*)(ws + ((size_t)120 << 20));     // 8MB
  __bf16* qb    = (__bf16*)(ws + ((size_t)128 << 20));     // 32MB (Q roped)
  __bf16* keys  = (__bf16*)(ws + ((size_t)160 << 20));     // 16MB
  __bf16* vals  = (__bf16*)(ws + ((size_t)176 << 20));     // 16MB
  __bf16* wob   = wqb;      // reuse after Q projection
  __bf16* attnb = xb;       // reuse after projections

  const int M = BSZ * SEQ;  // 4096 token rows

  // one-shot f32 -> bf16 downcasts
  cvt_f32_bf16_kernel<<<(M * DMODEL / 4 + 255) / 256, 256, 0, stream>>>(x,  xb,  M * DMODEL / 4);
  cvt_f32_bf16_kernel<<<(DMODEL * DMODEL / 4 + 255) / 256, 256, 0, stream>>>(wq, wqb, DMODEL * DMODEL / 4);
  cvt_f32_bf16_kernel<<<(NKV * HD * DMODEL / 4 + 255) / 256, 256, 0, stream>>>(wk, wkb, NKV * HD * DMODEL / 4);
  cvt_f32_bf16_kernel<<<(NKV * HD * DMODEL / 4 + 255) / 256, 256, 0, stream>>>(wv, wvb, NKV * HD * DMODEL / 4);

  // projections (bf16 in, bf16 out, f32 accumulate)
  gemm_bf16<DMODEL, DMODEL, true><<<dim3(DMODEL / 128, M / 128), 256, 0, stream>>>(xb, wqb, qraw);
  // wq no longer needed -> convert wo into its slot (stream-ordered after GEMM above)
  cvt_f32_bf16_kernel<<<(DMODEL * DMODEL / 4 + 255) / 256, 256, 0, stream>>>(wo, wob, DMODEL * DMODEL / 4);
  gemm_bf16<NKV * HD, DMODEL, true><<<dim3(NKV * HD / 128, M / 128), 256, 0, stream>>>(xb, wkb, kraw);
  gemm_bf16<NKV * HD, DMODEL, true><<<dim3(NKV * HD / 128, M / 128), 256, 0, stream>>>(xb, wvb, vraw);

  // RoPE + KV-cache merge + layout transforms
  rope_q_kernel<<<(BSZ * NH * SEQ * 64) / 256, 256, 0, stream>>>(qraw, fcos, fsin, qb);
  rope_k_kernel<<<(BSZ * NKV * SEQ * 64) / 256, 256, 0, stream>>>(kraw, fcos, fsin, keys);
  cache_k_kernel<<<(BSZ * NKV * PASTN * HD) / 256, 256, 0, stream>>>(ck, keys);
  v_new_kernel<<<(BSZ * SEQ * NKV * HD) / 256, 256, 0, stream>>>(vraw, vals);
  cache_v_kernel<<<(BSZ * NKV * PASTN * HD) / 256, 256, 0, stream>>>(cv, vals);

  // flash attention (8192 wave-tiles, 4 waves/block) -> bf16 token-major
  attn_kernel<<<(BSZ * NH * (SEQ / 16)) / 4, 128, 0, stream>>>(qb, keys, vals, attnb);

  // output projection -> f32 d_out
  gemm_bf16<DMODEL, DMODEL, false><<<dim3(DMODEL / 128, M / 128), 256, 0, stream>>>(attnb, wob, out);
}
